// MovieUserEmbeddingModel_58969900974303
// MI455X (gfx1250) — matrix operand order repaired and checked by reading, not compiled, not run on hardware
//
#include <hip/hip_runtime.h>
#include <hip/hip_bf16.h>

typedef __attribute__((ext_vector_type(16))) __bf16 v16bf;
typedef __attribute__((ext_vector_type(8)))  __bf16 v8bf;
typedef __attribute__((ext_vector_type(8)))  float  v8f;

#define NUM_MOVIES 100000
#define NUM_USERS  50000
#define N_NODES    (NUM_MOVIES + NUM_USERS)   // 150000, divisible by 16
#define EMB        128
#define HID        256
#define NEDGES     500000

// ---------------- utility kernels ----------------

__global__ void zero_f32(float* __restrict__ p, size_t n) {
    size_t i = (size_t)blockIdx.x * blockDim.x + threadIdx.x;
    size_t st = (size_t)gridDim.x * blockDim.x;
    for (; i < n; i += st) p[i] = 0.0f;
}

__global__ void deg_kernel(const long long* __restrict__ ei, float* __restrict__ deg, int E) {
    int e = blockIdx.x * blockDim.x + threadIdx.x;
    if (e < E) {
        int dst = (int)ei[(size_t)E + e];
        atomicAdd(deg + dst, 1.0f);
    }
}

// x = concat(movie_emb, user_emb) converted to bf16 row-major [N, EMB]
__global__ void concat_to_bf16(const float* __restrict__ movie,
                               const float* __restrict__ user,
                               __bf16* __restrict__ xbf) {
    size_t i = (size_t)blockIdx.x * blockDim.x + threadIdx.x;
    size_t st = (size_t)gridDim.x * blockDim.x;
    const size_t n = (size_t)N_NODES * EMB;
    const size_t moff = (size_t)NUM_MOVIES * EMB;
    for (; i < n; i += st) {
        float v = (i < moff) ? movie[i] : user[i - moff];
        xbf[i] = (__bf16)v;
    }
}

// layer-1 scatter: agg[dst, :] += x[src, :]   (f32 atomics, agg is L2-resident)
__global__ void scatter_emb(const long long* __restrict__ ei,
                            const float* __restrict__ movie,
                            const float* __restrict__ user,
                            float* __restrict__ agg, int E) {
    int e = blockIdx.x;
    int k = threadIdx.x;           // 0..127
    int src = (int)ei[e];
    int dst = (int)ei[(size_t)E + e];
    const float* xr = (src < NUM_MOVIES)
                      ? (movie + (size_t)src * EMB)
                      : (user  + (size_t)(src - NUM_MOVIES) * EMB);
    atomicAdd(&agg[(size_t)dst * EMB + k], xr[k]);
}

// layer-2 scatter: agg[dst, :] += h[src, :].
// h stored bf16; each of 128 threads loads one dword (2 bf16) -> coalesced
// full-cacheline reads, then two f32 atomics into L2-resident agg.
__global__ void scatter_h(const long long* __restrict__ ei,
                          const __bf16* __restrict__ hbf,
                          float* __restrict__ agg, int E) {
    int e = blockIdx.x;
    int t = threadIdx.x;           // 0..127, covers 2 elems each
    int src = (int)ei[e];
    int dst = (int)ei[(size_t)E + e];
    const __bf16* hrow = hbf + (size_t)src * HID;
    float2 v;
    {   // one b32 load, unpack two bf16
        const ushort2 u = ((const ushort2*)hrow)[t];
        union { unsigned u; float f; } lo, hi;
        lo.u = (unsigned)u.x << 16;
        hi.u = (unsigned)u.y << 16;
        v.x = lo.f; v.y = hi.f;
    }
    float* arow = agg + (size_t)dst * HID + 2 * t;
    atomicAdd(arow,     v.x);
    atomicAdd(arow + 1, v.y);
}

// out_bf[n,k] = agg[n,k] / max(deg[n],1) ; K = 1<<shift
__global__ void normalize_to_bf16(const float* __restrict__ agg,
                                  const float* __restrict__ deg,
                                  __bf16* __restrict__ out,
                                  int shift, size_t n) {
    size_t i = (size_t)blockIdx.x * blockDim.x + threadIdx.x;
    size_t st = (size_t)gridDim.x * blockDim.x;
    for (; i < n; i += st) {
        float d = deg[i >> shift];
        out[i] = (__bf16)(agg[i] * (1.0f / fmaxf(d, 1.0f)));
    }
}

// ---------------- weight prepack: WMMA-fragment-major bf16 ----------------
// W: [K, Nout] f32 row-major -> Bp[tileN][kb][lane][16] bf16, so in the GEMM
// each lane reads its v16bf fragment as one contiguous 32B chunk.
__global__ void pack_w(const float* __restrict__ W, __bf16* __restrict__ Bp,
                       int K, int Nout) {
    int tile = blockIdx.x;                // Nout/16 tiles
    int kb   = blockIdx.y;                // K/32 blocks
    int lane = threadIdx.x;               // 32
    int half = lane >> 4;
    int r    = lane & 15;
    int col  = tile * 16 + r;
    __bf16* out = Bp + (((size_t)tile * (K / 32) + kb) * 32 + lane) * 16;
    #pragma unroll
    for (int j = 0; j < 16; ++j) {
        int k = kb * 32 + (j < 8 ? half * 8 + j : 16 + half * 8 + (j - 8));  // 16-bit B 32x16: elem j -> K=8*half+j (j<8), 16+8*half+(j-8) (j>=8)
        out[j] = (__bf16)W[(size_t)k * Nout + col];
    }
}

// ---------------- fused SAGE dense layer via bf16 WMMA ----------------
// out[M, Nout] = (relu?)( Aagg @ Wl + bias + X @ Wr ),  A inputs bf16 row-major,
// B prepacked fragments. One wave computes a 16x64 strip (4 accumulators).
// Per K-step: A = 4x b128, all 8 B fragments preloaded (16x b128 in one clause),
// then 8 back-to-back v_wmma behind a single load-wait. K templated -> full unroll.
template <int K, int RELU>
__global__ __launch_bounds__(32) void sage_gemm_bf16(
    const __bf16* __restrict__ Aagg, const __bf16* __restrict__ X,
    const __bf16* __restrict__ Bpl, const __bf16* __restrict__ Bpr,
    const float* __restrict__ bias,
    float* __restrict__ outf, __bf16* __restrict__ outbf, int Nout)
{
    const int lane = threadIdx.x;       // 0..31
    const int half = lane >> 4;         // 0 or 1
    const int r    = lane & 15;
    const int tileM = blockIdx.x;
    const int tg    = blockIdx.y;       // group of 4 N-tiles
    const int row = tileM * 16 + r;
    constexpr int K32 = K / 32;

    const __bf16* arow = Aagg + (size_t)row * K;
    const __bf16* xrow = X    + (size_t)row * K;

    v8f acc[4] = {{}, {}, {}, {}};

    #pragma unroll
    for (int kb = 0; kb < K32; ++kb) {
        const int k0 = kb * 32;
        // 16-bit A 16x32 fragment: two contiguous 8-elem (16B) chunks per lane
        v8bf alo = *(const v8bf*)(arow + k0 + half * 8);
        v8bf ahi = *(const v8bf*)(arow + k0 + half * 8 + 16);
        v8bf xlo = *(const v8bf*)(xrow + k0 + half * 8);
        v8bf xhi = *(const v8bf*)(xrow + k0 + half * 8 + 16);
        if (kb + 1 < K32) {  // prefetch next K-step of the streaming A operands
            __builtin_prefetch(arow + k0 + 32, 0, 1);
            __builtin_prefetch(xrow + k0 + 32, 0, 1);
        }
        v16bf a1, a2;
        #pragma unroll
        for (int j = 0; j < 8; ++j) {
            a1[j] = alo[j]; a1[j + 8] = ahi[j];
            a2[j] = xlo[j]; a2[j + 8] = xhi[j];
        }
        // preload ALL 8 B fragments for this K-step (one big load clause)
        v16bf bl[4], br[4];
        #pragma unroll
        for (int t = 0; t < 4; ++t) {
            const size_t foff =
                (((size_t)(tg * 4 + t) * K32 + kb) * 32 + lane) * 16;
            bl[t] = *(const v16bf*)(Bpl + foff);
            br[t] = *(const v16bf*)(Bpr + foff);
        }
        // 8 back-to-back WMMAs into 4 independent accumulators
        #pragma unroll
        for (int t = 0; t < 4; ++t) {
            acc[t] = __builtin_amdgcn_wmma_f32_16x16x32_bf16(
                false, a1, false, bl[t], (short)0, acc[t], false, false);
            acc[t] = __builtin_amdgcn_wmma_f32_16x16x32_bf16(
                false, a2, false, br[t], (short)0, acc[t], false, false);
        }
    }

    // C/D layout: VGPR v holds M = v + 8*half, N = lane&15
    #pragma unroll
    for (int t = 0; t < 4; ++t) {
        const int col = (tg * 4 + t) * 16 + r;
        const float bv = bias[col];
        #pragma unroll
        for (int v = 0; v < 8; ++v) {
            int m = tileM * 16 + v + 8 * half;
            float val = acc[t][v] + bv;
            if (RELU) val = fmaxf(val, 0.0f);
            if (outf)  outf[(size_t)m * Nout + col]  = val;
            if (outbf) outbf[(size_t)m * Nout + col] = (__bf16)val;
        }
    }
}

// ---------------- edge scoring: wave-per-edge dot product ----------------
__global__ void edge_score(const long long* __restrict__ ei,
                           const float* __restrict__ refined,
                           float* __restrict__ ratings, int E) {
    int widx = (int)(((size_t)blockIdx.x * blockDim.x + threadIdx.x) >> 5);
    int lane = threadIdx.x & 31;
    if (widx >= E) return;
    int src = (int)ei[widx];
    int dst = (int)ei[(size_t)E + widx];
    const float4* a = (const float4*)(refined + (size_t)src * EMB);
    const float4* b = (const float4*)(refined + (size_t)dst * EMB);
    float4 va = a[lane];
    float4 vb = b[lane];
    float s = va.x * vb.x + va.y * vb.y + va.z * vb.z + va.w * vb.w;
    s += __shfl_xor(s, 16, 32);
    s += __shfl_xor(s,  8, 32);
    s += __shfl_xor(s,  4, 32);
    s += __shfl_xor(s,  2, 32);
    s += __shfl_xor(s,  1, 32);
    if (lane == 0) ratings[widx] = s;
}

// ---------------- launcher ----------------

extern "C" void kernel_launch(void* const* d_in, const int* in_sizes, int n_in,
                              void* d_out, int out_size, void* d_ws, size_t ws_size,
                              hipStream_t stream) {
    const long long* ei = (const long long*)d_in[0];  // [2, E] int64
    // d_in[1] = edge_attr (unused by reference)
    const float* movie = (const float*)d_in[2];  // [100000,128]
    const float* user  = (const float*)d_in[3];  // [50000,128]
    const float* Wl1   = (const float*)d_in[4];  // [128,256]
    const float* bl1   = (const float*)d_in[5];  // [256]
    const float* Wr1   = (const float*)d_in[6];  // [128,256]
    const float* Wl2   = (const float*)d_in[7];  // [256,128]
    const float* bl2   = (const float*)d_in[8];  // [128]
    const float* Wr2   = (const float*)d_in[9];  // [256,128]

    const int E = NEDGES;
    float* ratings = (float*)d_out;          // [E]
    float* refined = (float*)d_out + E;      // [N, 128] f32

    // ---- workspace layout ----
    // f32: deg[pad N] | agg[N*256]
    // bf16: xbf[N*128] | hbf[N*256] | aggbf[N*256] | 4 packed-W buffers
    float* deg = (float*)d_ws;
    float* agg = deg + ((N_NODES + 63) & ~63);
    __bf16* xbf   = (__bf16*)(agg + (size_t)N_NODES * HID);
    __bf16* hbf   = xbf + (size_t)N_NODES * EMB;
    __bf16* aggbf = hbf + (size_t)N_NODES * HID;
    const size_t BP1 = (size_t)(HID / 16) * (EMB / 32) * 32 * 16; // 32768
    const size_t BP2 = (size_t)(EMB / 16) * (HID / 32) * 32 * 16; // 32768
    __bf16* Bp1l = aggbf + (size_t)N_NODES * HID;
    __bf16* Bp1r = Bp1l + BP1;
    __bf16* Bp2l = Bp1r + BP1;
    __bf16* Bp2r = Bp2l + BP2;

    // ---- one-time prep (cheap): weight prepack + x -> bf16 ----
    pack_w<<<dim3(HID / 16, EMB / 32), 32, 0, stream>>>(Wl1, Bp1l, EMB, HID);
    pack_w<<<dim3(HID / 16, EMB / 32), 32, 0, stream>>>(Wr1, Bp1r, EMB, HID);
    pack_w<<<dim3(EMB / 16, HID / 32), 32, 0, stream>>>(Wl2, Bp2l, HID, EMB);
    pack_w<<<dim3(EMB / 16, HID / 32), 32, 0, stream>>>(Wr2, Bp2r, HID, EMB);
    concat_to_bf16<<<8192, 256, 0, stream>>>(movie, user, xbf);

    // ---- layer 1 ----
    zero_f32<<<1024, 256, 0, stream>>>(deg, (size_t)N_NODES);
    zero_f32<<<8192, 256, 0, stream>>>(agg, (size_t)N_NODES * EMB);
    deg_kernel<<<(E + 255) / 256, 256, 0, stream>>>(ei, deg, E);
    scatter_emb<<<E, EMB, 0, stream>>>(ei, movie, user, agg, E);
    normalize_to_bf16<<<8192, 256, 0, stream>>>(agg, deg, aggbf, 7,
                                                (size_t)N_NODES * EMB);
    sage_gemm_bf16<EMB, 1><<<dim3(N_NODES / 16, HID / 64), 32, 0, stream>>>(
        aggbf, xbf, Bp1l, Bp1r, bl1, nullptr, hbf, HID);

    // ---- layer 2 ----
    zero_f32<<<8192, 256, 0, stream>>>(agg, (size_t)N_NODES * HID);
    scatter_h<<<E, HID / 2, 0, stream>>>(ei, hbf, agg, E);
    normalize_to_bf16<<<8192, 256, 0, stream>>>(agg, deg, aggbf, 8,
                                                (size_t)N_NODES * HID);
    sage_gemm_bf16<HID, 0><<<dim3(N_NODES / 16, EMB / 64), 32, 0, stream>>>(
        aggbf, hbf, Bp2l, Bp2r, bl2, refined, nullptr, EMB);

    // ---- edge-wise scoring ----
    edge_score<<<(E * 32 + 255) / 256, 256, 0, stream>>>(ei, refined, ratings, E);
}
